// TransformerEncoderLayerWithCache_84963043049655
// MI455X (gfx1250) — compile-verified, matrix operand order checked by reading, not run on hardware
//
#include <hip/hip_runtime.h>
#include <stdint.h>

// Problem constants (fixed by the reference)
#define BQ   4
#define SEQ  1024
#define PP   1024
#define DD   1024
#define HH   16
#define HDIM 64
#define DFFN 4096
#define TT   2048   // P + S

typedef __attribute__((ext_vector_type(16))) __bf16 v16bf;
typedef __attribute__((ext_vector_type(8)))  float  v8f;
typedef __attribute__((ext_vector_type(4)))  unsigned int v4u;
typedef __attribute__((ext_vector_type(8)))  int v8i;
typedef __attribute__((ext_vector_type(4)))  int v4i;

union Frag16 { v16bf v; unsigned short u[16]; };

enum GemmMode { MODE_F32 = 0, MODE_RELU_BF16 = 2, MODE_RESID_F32 = 3, MODE_QKV = 4 };

#if defined(__has_builtin)
#  if __has_builtin(__builtin_amdgcn_tensor_load_to_lds) && __has_builtin(__builtin_amdgcn_s_wait_tensorcnt)
#    define HAVE_TDM 1
#  else
#    define HAVE_TDM 0
#  endif
#else
#  define HAVE_TDM 0
#endif

__device__ __forceinline__ unsigned short f2bf(float f) {
  union { float f; unsigned int u; } c; c.f = f;
  unsigned int u = c.u;
  unsigned int r = u + 0x7FFFu + ((u >> 16) & 1u);  // round-to-nearest-even
  return (unsigned short)(r >> 16);
}

#if HAVE_TDM
// ---------------------------------------------------------------------------
// Tensor Data Mover: DMA a 2D tile of bf16 (data_size=2B) from global into LDS.
// Descriptor bits per CDNA5 ISA D# group0/group1 (tile_w x tile_h, row stride
// in elements). Issued once per wave; completion tracked via TENSORcnt.
// ---------------------------------------------------------------------------
__device__ __forceinline__ void tdm_load_tile_bf16(unsigned lds_addr, const void* gptr,
                                                   int tile_w, int tile_h,
                                                   long long stride_elems) {
  unsigned long long ga = (unsigned long long)(size_t)gptr;
  v4u g0;
  g0[0] = 1u;                                            // count=1, is_restore=0
  g0[1] = lds_addr;                                      // LDS byte address
  g0[2] = (unsigned)(ga & 0xFFFFFFFFu);                  // global_addr[31:0]
  g0[3] = (unsigned)((ga >> 32) & 0x01FFFFFFu) | (2u << 30);  // global_addr[56:32], type=2
  v8i g1;
  g1[0] = (1 << 16);                                     // data_size=1 (2 bytes), no pad/iter
  g1[1] = (int)(((unsigned)tile_w & 0xFFFFu) << 16);     // tensor_dim0[15:0]  (== tile_w)
  g1[2] = (int)((((unsigned)tile_w >> 16) & 0xFFFFu)) |
          (int)(((unsigned)tile_h & 0xFFFFu) << 16);     // tensor_dim0[31:16] | tensor_dim1[15:0]
  g1[3] = (int)((((unsigned)tile_h >> 16) & 0xFFFFu)) |
          (int)(((unsigned)tile_w & 0xFFFFu) << 16);     // tensor_dim1[31:16] | tile_dim0
  g1[4] = (int)((unsigned)tile_h & 0xFFFFu);             // tile_dim1 (tile_dim2 = 0)
  g1[5] = (int)(unsigned)(stride_elems & 0xFFFFFFFFll);  // tensor_dim0_stride[31:0]
  g1[6] = (int)(unsigned)((stride_elems >> 32) & 0xFFFFll); // stride[47:32]; dim1_stride=0
  g1[7] = 0;
  v4i z = {0, 0, 0, 0};
#if __clang_major__ >= 23
  v8i z8 = {0, 0, 0, 0, 0, 0, 0, 0};
  __builtin_amdgcn_tensor_load_to_lds(g0, g1, z, z, z8, 0);
#else
  __builtin_amdgcn_tensor_load_to_lds(g0, g1, z, z, 0);
#endif
}
#endif

// ---------------------------------------------------------------------------
// f32 -> bf16 elementwise conversion (weights)
// ---------------------------------------------------------------------------
__global__ void cvt_bf16_k(const float* __restrict__ in, unsigned short* __restrict__ out, int n) {
  int i = blockIdx.x * blockDim.x + threadIdx.x;
  int stride = gridDim.x * blockDim.x;
  for (; i < n; i += stride) out[i] = f2bf(in[i]);
}

// ---------------------------------------------------------------------------
// LayerNorm over D=1024, one block (256 threads) per row, output bf16
// ---------------------------------------------------------------------------
__global__ __launch_bounds__(256)
void layernorm_k(const float* __restrict__ x, const float* __restrict__ g,
                 const float* __restrict__ bb, unsigned short* __restrict__ out) {
  __shared__ float s1[256];
  __shared__ float s2[256];
  const int row = blockIdx.x;
  const int tid = threadIdx.x;
  const float* xr = x + (size_t)row * DD;
  float v[4], a = 0.f, q = 0.f;
#pragma unroll
  for (int i = 0; i < 4; ++i) { v[i] = xr[tid + i * 256]; a += v[i]; q += v[i] * v[i]; }
  s1[tid] = a; s2[tid] = q;
  __syncthreads();
  for (int st = 128; st > 0; st >>= 1) {
    if (tid < st) { s1[tid] += s1[tid + st]; s2[tid] += s2[tid + st]; }
    __syncthreads();
  }
  const float mean = s1[0] * (1.0f / DD);
  const float var  = s2[0] * (1.0f / DD) - mean * mean;
  const float rst  = rsqrtf(var + 1e-5f);
#pragma unroll
  for (int i = 0; i < 4; ++i) {
    const int c = tid + i * 256;
    out[(size_t)row * DD + c] = f2bf((v[i] - mean) * rst * g[c] + bb[c]);
  }
}

// ---------------------------------------------------------------------------
// Copy past_k/past_v [B,H,P,HD] into the full cache [B,H,T,HD] (f32 out + bf16 ws)
// ---------------------------------------------------------------------------
__global__ void past_copy_k(const float* __restrict__ past, float* __restrict__ outF,
                            unsigned short* __restrict__ outB) {
  const int n = BQ * HH * PP * HDIM;
  int i = blockIdx.x * blockDim.x + threadIdx.x;
  int stride = gridDim.x * blockDim.x;
  for (; i < n; i += stride) {
    int hd = i & (HDIM - 1);
    int p  = (i >> 6) & (PP - 1);
    int bh = i >> 16;
    size_t dst = ((size_t)bh * TT + p) * HDIM + hd;
    float val = past[i];
    outF[dst] = val;
    outB[dst] = f2bf(val);
  }
}

// ---------------------------------------------------------------------------
// Tiled WMMA bf16 GEMM: C[M,N] = A[M,K] @ B[K,N], fused epilogues.
// 128x128x32 tiles, 256 threads (8 waves), wave w -> rows [16w,16w+16), 8 N-tiles.
// A tile staged via TDM (tensor_load_to_lds) while all threads transpose-stage B.
// ---------------------------------------------------------------------------
#define GBM 128
#define GBN 128
#define GBK 32

__global__ __launch_bounds__(256)
void gemm_bf16_k(const unsigned short* __restrict__ A,
                 const unsigned short* __restrict__ Bw,
                 const float* __restrict__ bias,
                 const float* __restrict__ resid,
                 float* __restrict__ outF,
                 unsigned short* __restrict__ outB,
                 int M, int N, int K, int mode, int Tcap, int toff) {
  __shared__ __align__(16) unsigned short As[GBM][GBK];
  __shared__ __align__(16) unsigned short Bs[GBN][GBK];  // stored transposed: Bs[n][k]

  const int tid  = threadIdx.x;
  const int lane = tid & 31;
  const int wave = tid >> 5;
  const int half = lane >> 4;
  const int lq   = lane & 15;
  const int kk0  = half * 8;          // bf16 A/B lane K-group base: {0-7,16-23} / {8-15,24-31}

  const int m0 = blockIdx.y * GBM;
  const int n0 = blockIdx.x * GBN;

  v8f acc[8];
#pragma unroll
  for (int t = 0; t < 8; ++t)
#pragma unroll
    for (int r = 0; r < 8; ++r) acc[t][r] = 0.0f;

  const int arow = wave * 16 + lq;
#if HAVE_TDM
  const unsigned asLds = (unsigned)(size_t)(&As[0][0]);  // LDS offset (low 32 bits of flat)
#endif

  for (int kp = 0; kp < K; kp += GBK) {
#if HAVE_TDM
    // ---- stage A tile (128 x 32) via Tensor Data Mover (wave 0 issues DMA)
    if (tid < 32) {
      tdm_load_tile_bf16(asLds, A + (size_t)m0 * K + kp, GBK, GBM, (long long)K);
    }
#else
    // ---- fallback: stage A tile with coalesced 16B vector loads
    {
      int row = tid >> 1;
      int kc  = (tid & 1) * 16;
      const uint4* src = (const uint4*)(A + (size_t)(m0 + row) * K + kp + kc);
      uint4 d0 = src[0];
      uint4 d1 = src[1];
      *(uint4*)&As[row][kc]     = d0;
      *(uint4*)&As[row][kc + 8] = d1;
    }
#endif
    // ---- stage B tile (32 x 128) transposed into Bs[n][k] (all 256 threads)
    {
      int kk = tid >> 3;
      int nc = (tid & 7) * 16;
      union { uint4 q[2]; unsigned short s[16]; } tmp;
      const uint4* src = (const uint4*)(Bw + (size_t)(kp + kk) * N + n0 + nc);
      tmp.q[0] = src[0];
      tmp.q[1] = src[1];
#pragma unroll
      for (int i = 0; i < 16; ++i) Bs[nc + i][kk] = tmp.s[i];
      if (kp + GBK < K)
        __builtin_prefetch(Bw + (size_t)(kp + GBK + kk) * N + n0 + nc, 0, 3);
    }
#if HAVE_TDM
    if (tid < 32) __builtin_amdgcn_s_wait_tensorcnt(0);   // A tile landed in LDS
#endif
    __syncthreads();

    // ---- A fragment (shared by all 8 N-tiles of this wave)
    Frag16 a;
#pragma unroll
    for (int i = 0; i < 8; ++i) {
      a.u[i]     = As[arow][kk0 + i];
      a.u[8 + i] = As[arow][kk0 + 16 + i];
    }
#pragma unroll
    for (int nt = 0; nt < 8; ++nt) {
      Frag16 bfr;
      const int bn = nt * 16 + lq;
#pragma unroll
      for (int i = 0; i < 8; ++i) {
        bfr.u[i]     = Bs[bn][kk0 + i];
        bfr.u[8 + i] = Bs[bn][kk0 + 16 + i];
      }
      acc[nt] = __builtin_amdgcn_wmma_f32_16x16x32_bf16(false, a.v, false, bfr.v,
                                                        (short)0, acc[nt], false, false);
    }
    __syncthreads();
  }

  // ---- fused epilogue (C layout: VGPR r -> row r+8*half, col = lq within tile)
#pragma unroll
  for (int nt = 0; nt < 8; ++nt) {
#pragma unroll
    for (int r = 0; r < 8; ++r) {
      float c = acc[nt][r];
      const int m = m0 + wave * 16 + r + 8 * half;
      const int n = n0 + nt * 16 + lq;
      if (mode == MODE_RESID_F32) {
        float vv = c + (bias ? bias[n] : 0.0f) + resid[(size_t)m * N + n];
        outF[(size_t)m * N + n] = vv;
      } else if (mode == MODE_RELU_BF16) {
        float vv = c + bias[n];
        outB[(size_t)m * N + n] = f2bf(vv > 0.0f ? vv : 0.0f);
      } else if (mode == MODE_QKV) {
        const int b  = m >> 10;          // m / SEQ
        const int s  = m & (SEQ - 1);
        const int h  = n >> 6;           // n / HDIM
        const int hd = n & (HDIM - 1);
        size_t idx = (((size_t)(b * HH + h) * Tcap) + toff + s) * HDIM + hd;
        outB[idx] = f2bf(c);
        if (outF) outF[idx] = c;         // fp32 KV-cache output
      } else {
        outF[(size_t)m * N + n] = c;
      }
    }
  }
}

// ---------------------------------------------------------------------------
// Flash attention: one wave per (b, h, 16-query tile); causal with offset P.
// scores = q k^T / 8, online softmax, o = P v. bf16 WMMA, f32 stats.
// ---------------------------------------------------------------------------
__global__ __launch_bounds__(32)
void attn_k(const unsigned short* __restrict__ qb,
            const unsigned short* __restrict__ kb,
            const unsigned short* __restrict__ vb,
            unsigned short* __restrict__ out) {
  __shared__ __align__(16) unsigned short Pl[16][32];
  __shared__ __align__(16) unsigned short Vt[HDIM][32];   // V chunk transposed [hd][key]

  const int qt = blockIdx.x;
  const int h  = blockIdx.y;
  const int b  = blockIdx.z;
  const int qbase = qt * 16;

  const int lane = threadIdx.x;
  const int half = lane >> 4;
  const int lq   = lane & 15;
  const int kk0  = half * 8;

  const size_t bh = (size_t)(b * HH + h);
  const unsigned short* qrow    = qb + (bh * SEQ + qbase + lq) * HDIM;
  const unsigned short* kbase_p = kb + bh * TT * HDIM;
  const unsigned short* vbase_p = vb + bh * TT * HDIM;

  // Q fragments: K-dim = HDIM split into two 32-wide WMMA K chunks
  Frag16 qf0, qf1;
#pragma unroll
  for (int i = 0; i < 8; ++i) {
    qf0.u[i]     = qrow[kk0 + i];
    qf0.u[8 + i] = qrow[kk0 + 16 + i];
    qf1.u[i]     = qrow[32 + kk0 + i];
    qf1.u[8 + i] = qrow[32 + kk0 + 16 + i];
  }

  float mi[8], li[8];
  v8f o[4];
#pragma unroll
  for (int r = 0; r < 8; ++r) { mi[r] = -1.0e30f; li[r] = 0.0f; }
#pragma unroll
  for (int t = 0; t < 4; ++t)
#pragma unroll
    for (int r = 0; r < 8; ++r) o[t][r] = 0.0f;

  const int kmax   = qbase + 16 + PP;          // exclusive key bound for this tile
  const int nchunk = (kmax + 31) >> 5;         // 32 keys per chunk

  for (int kc = 0; kc < nchunk; ++kc) {
    const int j0 = kc * 32;

    // stage V chunk transposed: Vt[hd][key_local]
    {
      const unsigned short* vrow = vbase_p + (size_t)(j0 + lane) * HDIM;
#pragma unroll
      for (int i = 0; i < HDIM; ++i) Vt[i][lane] = vrow[i];
    }

    // ---- scores: two 16x16 tiles (keys j0..j0+15, j0+16..j0+31)
    v8f s0, s1;
#pragma unroll
    for (int r = 0; r < 8; ++r) { s0[r] = 0.0f; s1[r] = 0.0f; }
    {
      const unsigned short* kr0 = kbase_p + (size_t)(j0 + lq) * HDIM;
      const unsigned short* kr1 = kbase_p + (size_t)(j0 + 16 + lq) * HDIM;
      Frag16 k00, k01, k10, k11;
#pragma unroll
      for (int i = 0; i < 8; ++i) {
        k00.u[i] = kr0[kk0 + i];       k00.u[8 + i] = kr0[kk0 + 16 + i];
        k01.u[i] = kr0[32 + kk0 + i];  k01.u[8 + i] = kr0[32 + kk0 + 16 + i];
        k10.u[i] = kr1[kk0 + i];       k10.u[8 + i] = kr1[kk0 + 16 + i];
        k11.u[i] = kr1[32 + kk0 + i];  k11.u[8 + i] = kr1[32 + kk0 + 16 + i];
      }
      s0 = __builtin_amdgcn_wmma_f32_16x16x32_bf16(false, qf0.v, false, k00.v, (short)0, s0, false, false);
      s0 = __builtin_amdgcn_wmma_f32_16x16x32_bf16(false, qf1.v, false, k01.v, (short)0, s0, false, false);
      s1 = __builtin_amdgcn_wmma_f32_16x16x32_bf16(false, qf0.v, false, k10.v, (short)0, s1, false, false);
      s1 = __builtin_amdgcn_wmma_f32_16x16x32_bf16(false, qf1.v, false, k11.v, (short)0, s1, false, false);
    }

    // ---- online softmax (row M = r + 8*half; reductions across 16-lane half)
    float fct[8];
#pragma unroll
    for (int r = 0; r < 8; ++r) {
      const int qg = qbase + r + 8 * half;
      float a0 = s0[r] * 0.125f;                    // 1/sqrt(64)
      float a1 = s1[r] * 0.125f;
      if (j0 + lq      > qg + PP) a0 = -1.0e30f;    // causal: key j <= q + P
      if (j0 + 16 + lq > qg + PP) a1 = -1.0e30f;
      float vm = fmaxf(a0, a1);
#pragma unroll
      for (int off = 1; off < 16; off <<= 1) vm = fmaxf(vm, __shfl_xor(vm, off, 32));
      const float nm = fmaxf(mi[r], vm);
      const float fc = __expf(mi[r] - nm);
      const float p0 = __expf(a0 - nm);
      const float p1 = __expf(a1 - nm);
      float rs = p0 + p1;
#pragma unroll
      for (int off = 1; off < 16; off <<= 1) rs += __shfl_xor(rs, off, 32);
      li[r] = li[r] * fc + rs;
      mi[r] = nm;
      fct[r] = fc;
      Pl[r + 8 * half][lq]      = f2bf(p0);
      Pl[r + 8 * half][16 + lq] = f2bf(p1);
    }
#pragma unroll
    for (int t = 0; t < 4; ++t)
#pragma unroll
      for (int r = 0; r < 8; ++r) o[t][r] *= fct[r];

    // ---- o += P @ V (K = 32 keys, 4 N-tiles over HDIM)
    Frag16 pa;
#pragma unroll
    for (int i = 0; i < 8; ++i) {
      pa.u[i]     = Pl[lq][kk0 + i];
      pa.u[8 + i] = Pl[lq][kk0 + 16 + i];
    }
#pragma unroll
    for (int t = 0; t < 4; ++t) {
      Frag16 vf;
      const int vn = t * 16 + lq;
#pragma unroll
      for (int i = 0; i < 8; ++i) {
        vf.u[i]     = Vt[vn][kk0 + i];
        vf.u[8 + i] = Vt[vn][kk0 + 16 + i];
      }
      o[t] = __builtin_amdgcn_wmma_f32_16x16x32_bf16(false, pa.v, false, vf.v, (short)0, o[t], false, false);
    }
  }

  // ---- normalize & store attn output bf16 [B*S, D] (row = b*S+q, col = h*64+hd)
#pragma unroll
  for (int r = 0; r < 8; ++r) li[r] = 1.0f / li[r];
#pragma unroll
  for (int t = 0; t < 4; ++t)
#pragma unroll
    for (int r = 0; r < 8; ++r) {
      const size_t row = (size_t)b * SEQ + qbase + r + 8 * half;
      out[row * DD + h * HDIM + t * 16 + lq] = f2bf(o[t][r] * li[r]);
    }
}

// ---------------------------------------------------------------------------
// Host launcher
// ---------------------------------------------------------------------------
extern "C" void kernel_launch(void* const* d_in, const int* in_sizes, int n_in,
                              void* d_out, int out_size, void* d_ws, size_t ws_size,
                              hipStream_t stream) {
  (void)in_sizes; (void)n_in; (void)out_size; (void)ws_size;

  const float* x      = (const float*)d_in[0];
  const float* past_k = (const float*)d_in[1];
  const float* past_v = (const float*)d_in[2];
  const float* wq  = (const float*)d_in[3];
  const float* wk  = (const float*)d_in[4];
  const float* wv  = (const float*)d_in[5];
  const float* wo  = (const float*)d_in[6];
  const float* g1  = (const float*)d_in[7];
  const float* b1  = (const float*)d_in[8];
  const float* w1  = (const float*)d_in[9];
  const float* bb1 = (const float*)d_in[10];
  const float* w2  = (const float*)d_in[11];
  const float* bb2 = (const float*)d_in[12];
  const float* g2  = (const float*)d_in[13];
  const float* b2  = (const float*)d_in[14];

  float* outX = (float*)d_out;
  float* outK = outX + (size_t)BQ * SEQ * DD;
  float* outV = outK + (size_t)BQ * HH * TT * HDIM;

  char* ws = (char*)d_ws;
  size_t off = 0;
  auto alloc = [&](size_t bytes) -> void* {
    void* p = ws + off;
    off = (off + bytes + 255) & ~(size_t)255;
    return p;
  };
  const size_t MROW = (size_t)BQ * SEQ;  // 4096

  unsigned short* hB   = (unsigned short*)alloc(MROW * DD * 2);
  unsigned short* wqB  = (unsigned short*)alloc((size_t)DD * DD * 2);
  unsigned short* wkB  = (unsigned short*)alloc((size_t)DD * DD * 2);
  unsigned short* wvB  = (unsigned short*)alloc((size_t)DD * DD * 2);
  unsigned short* woB  = (unsigned short*)alloc((size_t)DD * DD * 2);
  unsigned short* w1B  = (unsigned short*)alloc((size_t)DD * DFFN * 2);
  unsigned short* w2B  = (unsigned short*)alloc((size_t)DFFN * DD * 2);
  unsigned short* qB   = (unsigned short*)alloc(MROW * DD * 2);
  unsigned short* kB   = (unsigned short*)alloc((size_t)BQ * HH * TT * HDIM * 2);
  unsigned short* vB   = (unsigned short*)alloc((size_t)BQ * HH * TT * HDIM * 2);
  unsigned short* attnB= (unsigned short*)alloc(MROW * DD * 2);
  float*          x1   = (float*)alloc(MROW * DD * 4);
  unsigned short* h2B  = (unsigned short*)alloc(MROW * DD * 2);
  unsigned short* ffB  = (unsigned short*)alloc(MROW * DFFN * 2);

  // 1) weights -> bf16
  cvt_bf16_k<<<2048, 256, 0, stream>>>(wq, wqB, DD * DD);
  cvt_bf16_k<<<2048, 256, 0, stream>>>(wk, wkB, DD * DD);
  cvt_bf16_k<<<2048, 256, 0, stream>>>(wv, wvB, DD * DD);
  cvt_bf16_k<<<2048, 256, 0, stream>>>(wo, woB, DD * DD);
  cvt_bf16_k<<<4096, 256, 0, stream>>>(w1, w1B, DD * DFFN);
  cvt_bf16_k<<<4096, 256, 0, stream>>>(w2, w2B, DFFN * DD);

  // 2) h = LN1(x) -> bf16
  layernorm_k<<<(unsigned)MROW, 256, 0, stream>>>(x, g1, b1, hB);

  // 3) QKV projections (WMMA + TDM staging), scattered to [B,H,T,HD]
  dim3 gqkv(DD / GBN, (unsigned)(MROW / GBM));
  gemm_bf16_k<<<gqkv, 256, 0, stream>>>(hB, wqB, nullptr, nullptr, nullptr, qB,
                                        (int)MROW, DD, DD, MODE_QKV, SEQ, 0);
  gemm_bf16_k<<<gqkv, 256, 0, stream>>>(hB, wkB, nullptr, nullptr, outK, kB,
                                        (int)MROW, DD, DD, MODE_QKV, TT, PP);
  gemm_bf16_k<<<gqkv, 256, 0, stream>>>(hB, wvB, nullptr, nullptr, outV, vB,
                                        (int)MROW, DD, DD, MODE_QKV, TT, PP);

  // 4) prepend past K/V (fp32 out + bf16 ws)
  past_copy_k<<<4096, 256, 0, stream>>>(past_k, outK, kB);
  past_copy_k<<<4096, 256, 0, stream>>>(past_v, outV, vB);

  // 5) flash attention (WMMA)
  attn_k<<<dim3(SEQ / 16, HH, BQ), 32, 0, stream>>>(qB, kB, vB, attnB);

  // 6) x1 = x + attn @ wo
  gemm_bf16_k<<<gqkv, 256, 0, stream>>>(attnB, woB, nullptr, x, x1, nullptr,
                                        (int)MROW, DD, DD, MODE_RESID_F32, 0, 0);

  // 7) h2 = LN2(x1) -> bf16
  layernorm_k<<<(unsigned)MROW, 256, 0, stream>>>(x1, g2, b2, h2B);

  // 8) ff1 = relu(h2 @ w1 + bb1) -> bf16
  dim3 gff1(DFFN / GBN, (unsigned)(MROW / GBM));
  gemm_bf16_k<<<gff1, 256, 0, stream>>>(h2B, w1B, bb1, nullptr, nullptr, ffB,
                                        (int)MROW, DFFN, DD, MODE_RELU_BF16, 0, 0);

  // 9) out_x = x1 + ff1 @ w2 + bb2  (fp32, final output)
  gemm_bf16_k<<<gqkv, 256, 0, stream>>>(ffB, w2B, bb2, x1, outX, nullptr,
                                        (int)MROW, DD, DFFN, MODE_RESID_F32, 0, 0);
}
